// MultiHeadAttention_20667382628419
// MI455X (gfx1250) — matrix operand order repaired and checked by reading, not compiled, not run on hardware
//
#include <hip/hip_runtime.h>

#define Bsz 2
#define Nsz 2048
#define Dsz 512
#define Hn  8
#define DKd 64
#define NEGV (-1e9f)
#define PSTR 40   // LDS P-tile row stride in halves (80B, 16B-aligned rows)

typedef __attribute__((ext_vector_type(16))) __bf16 v16bf;
typedef __attribute__((ext_vector_type(8)))  __bf16 v8bf;
typedef __attribute__((ext_vector_type(8)))  float  v8f;

static __device__ __forceinline__ v8f wmma_bf16(v16bf a, v16bf b, v8f c) {
  return __builtin_amdgcn_wmma_f32_16x16x32_bf16(false, a, false, b, (short)0, c, false, false);
}

// ---- A-operand loader: 16x32 (M x K), lane = row, K groups {0-7,16-23}/{8-15,24-31}
static __device__ __forceinline__ v16bf load_a_bf16(const __bf16* __restrict__ s, int ld, int lane) {
  const int m  = lane & 15;
  const int ko = (lane & 16) ? 8 : 0;
  const __bf16* r = s + (size_t)m * ld + ko;
  v8bf lo = *(const v8bf*)r;          // 16B-aligned
  v8bf hi = *(const v8bf*)(r + 16);
  v16bf a;
#pragma unroll
  for (int e = 0; e < 8; ++e) { a[e] = lo[e]; a[8 + e] = hi[e]; }
  return a;
}

// ---- B-operand loader: 32x16 (K x N), lane = column; source is B^T row-major.
// Lanes 0-15 hold K={0-7,16-23}, lanes 16-31 hold K={8-15,24-31} (matches A loader).
static __device__ __forceinline__ v16bf load_b_bf16(const __bf16* __restrict__ s, int ld, int lane) {
  const int n  = lane & 15;
  const int kb = (lane & 16) ? 8 : 0;
  const __bf16* r = s + (size_t)n * ld + kb;
  v8bf lo = *(const v8bf*)r;
  v8bf hi = *(const v8bf*)(r + 16);
  v16bf v;
#pragma unroll
  for (int e = 0; e < 8; ++e) { v[e] = lo[e]; v[8 + e] = hi[e]; }
  return v;
}

// ============================================================================
// Kernel A: bulk fp32 -> bf16 conversion (8 elements / thread, b128 in, b64+ out)
// ============================================================================
__global__ void cvt_bf16(const float* __restrict__ src, unsigned short* __restrict__ dst) {
  const size_t i = ((size_t)blockIdx.x * blockDim.x + threadIdx.x) * 8;
  v8bf o;
#pragma unroll
  for (int e = 0; e < 8; ++e) o[e] = (__bf16)src[i + e];
  *(v8bf*)((__bf16*)dst + i) = o;
}

// ============================================================================
// Kernel 0: combine seq_mask & sparse_mask into 1-bit-per-element words.
// packed[((b*2+p)*N + i)*(N/32) + w], bit j within word = column w*32+j.
// ============================================================================
__global__ void pack_masks(const int* __restrict__ seq_mask,
                           const int* __restrict__ sparse_masks,
                           unsigned* __restrict__ packed) {
  const int WPR = Nsz / 32;  // 64
  const int idx = blockIdx.x * blockDim.x + threadIdx.x;
  const int w = idx % WPR;
  const int i = (idx / WPR) % Nsz;
  const int p = (idx / (WPR * Nsz)) & 1;
  const int b = idx / (WPR * Nsz * 2);
  const int* sq = seq_mask + ((size_t)b * Nsz + i) * Nsz + w * 32;
  const int* sp = sparse_masks + ((size_t)p * Nsz + i) * Nsz + w * 32;
  unsigned m = 0;
#pragma unroll
  for (int j = 0; j < 32; ++j)
    m |= ((sq[j] != 0) && (sp[j] != 0)) ? (1u << j) : 0u;
  packed[idx] = m;
}

// ============================================================================
// Kernel 1: Q/K/V projections, Y = X @ W^T (bf16 WMMA, fp32 accum).
// One wave per 16x16 tile, dual accumulators to break WMMA D->C chains.
// Q,K -> [B,H,N,64] bf16; V -> [B,H,64,N] bf16 (transposed).
// ============================================================================
__global__ void qkv_proj(const unsigned short* __restrict__ x_p,
                         const unsigned short* __restrict__ w_p,   // Wq|Wk|Wv bf16
                         unsigned short* __restrict__ q_p, unsigned short* __restrict__ k_p,
                         unsigned short* __restrict__ vt_p) {
  const int lane = threadIdx.x & 31;
  const int wave = (blockIdx.x * blockDim.x + threadIdx.x) >> 5;
  const int jt  = wave & 31;
  const int it  = (wave >> 5) & 127;
  const int b   = (wave >> 12) & 1;
  const int mat = wave >> 13;  // 0=Q 1=K 2=V
  const __bf16* W  = (const __bf16*)w_p + (size_t)mat * Dsz * Dsz;
  const __bf16* xb = (const __bf16*)x_p + (size_t)b * Nsz * Dsz;
  const int i0 = it * 16, j0 = jt * 16;

  v8f c0 = {0.f,0.f,0.f,0.f,0.f,0.f,0.f,0.f};
  v8f c1 = {0.f,0.f,0.f,0.f,0.f,0.f,0.f,0.f};
#pragma unroll 2
  for (int k0 = 0; k0 < Dsz; k0 += 64) {
    c0 = wmma_bf16(load_a_bf16(xb + (size_t)i0 * Dsz + k0,      Dsz, lane),
                   load_b_bf16(W  + (size_t)j0 * Dsz + k0,      Dsz, lane), c0);
    c1 = wmma_bf16(load_a_bf16(xb + (size_t)i0 * Dsz + k0 + 32, Dsz, lane),
                   load_b_bf16(W  + (size_t)j0 * Dsz + k0 + 32, Dsz, lane), c1);
  }
#pragma unroll
  for (int r = 0; r < 8; ++r) c0[r] += c1[r];

  const int n  = lane & 15;
  const int mb = (lane & 16) ? 8 : 0;
  const int col = j0 + n;
  const int h = col >> 6, dd = col & 63;
  __bf16* qk = (__bf16*)((mat == 0) ? q_p : k_p);
  __bf16* vt = (__bf16*)vt_p;
#pragma unroll
  for (int r = 0; r < 8; ++r) {
    const int row = i0 + mb + r;
    const __bf16 v = (__bf16)c0[r];
    if (mat == 2)
      vt[((size_t)(b * Hn + h) * DKd + dd) * Nsz + row] = v;
    else
      qk[((size_t)(b * Hn + h) * Nsz + row) * DKd + dd] = v;
  }
}

// ============================================================================
// Kernel 2: flash attention. One wave per (b, h, 16-query-row tile).
// Online softmax, P staged through LDS for C->A operand re-layout.
// ============================================================================
__global__ void flash_attn(const unsigned short* __restrict__ q_p,
                           const unsigned short* __restrict__ k_p,
                           const unsigned short* __restrict__ vt_p,
                           const unsigned* __restrict__ packed,
                           unsigned short* __restrict__ o_p) {
  __shared__ __bf16 lds_p[8][16 * PSTR];
  const int lane  = threadIdx.x & 31;
  const int wslot = threadIdx.x >> 5;
  const int wave  = blockIdx.x * 8 + wslot;
  const int it = wave & 127;
  const int h  = (wave >> 7) & 7;
  const int b  = wave >> 10;
  const int i0 = it * 16;

  const __bf16* qb  = (const __bf16*)q_p  + (size_t)(b * Hn + h) * Nsz * DKd;
  const __bf16* kb  = (const __bf16*)k_p  + (size_t)(b * Hn + h) * Nsz * DKd;
  const __bf16* vtb = (const __bf16*)vt_p + (size_t)(b * Hn + h) * DKd * Nsz;
  const unsigned* pm = packed + (size_t)(b * 2 + (h & 1)) * Nsz * (Nsz / 32);

  // Q tile A-operands (dk=64 -> two K=32 operands), reused over whole row loop
  const v16bf qa0 = load_a_bf16(qb + (size_t)i0 * DKd + 0,  DKd, lane);
  const v16bf qa1 = load_a_bf16(qb + (size_t)i0 * DKd + 32, DKd, lane);

  v8f o0 = {0.f,0.f,0.f,0.f,0.f,0.f,0.f,0.f};
  v8f o1 = {0.f,0.f,0.f,0.f,0.f,0.f,0.f,0.f};
  v8f o2 = {0.f,0.f,0.f,0.f,0.f,0.f,0.f,0.f};
  v8f o3 = {0.f,0.f,0.f,0.f,0.f,0.f,0.f,0.f};
  float mrow[8], lrow[8];
#pragma unroll
  for (int r = 0; r < 8; ++r) { mrow[r] = -1e30f; lrow[r] = 0.f; }

  const int n  = lane & 15;
  const int mb = (lane & 16) ? 8 : 0;
  __bf16* myp = &lds_p[wslot][0];

  for (int j0 = 0; j0 < Nsz; j0 += 32) {
    // --- S = Q K^T for 16x32 key block (two independent 16x16 C tiles, interleaved)
    v8f s0 = {0.f,0.f,0.f,0.f,0.f,0.f,0.f,0.f};
    v8f s1 = {0.f,0.f,0.f,0.f,0.f,0.f,0.f,0.f};
    {
      const v16bf kb00 = load_b_bf16(kb + (size_t)j0 * DKd + 0,         DKd, lane);
      const v16bf kb01 = load_b_bf16(kb + (size_t)j0 * DKd + 32,        DKd, lane);
      const v16bf kb10 = load_b_bf16(kb + (size_t)(j0 + 16) * DKd + 0,  DKd, lane);
      const v16bf kb11 = load_b_bf16(kb + (size_t)(j0 + 16) * DKd + 32, DKd, lane);
      s0 = wmma_bf16(qa0, kb00, s0);
      s1 = wmma_bf16(qa0, kb10, s1);
      s0 = wmma_bf16(qa1, kb01, s0);
      s1 = wmma_bf16(qa1, kb11, s1);
    }

    if (j0 + 32 < Nsz) {
      __builtin_prefetch(kb + (size_t)(j0 + 32) * DKd, 0, 0);   // global_prefetch next K block
      __builtin_prefetch(vtb + (size_t)j0 + 32, 0, 0);
    }

    // --- scale + packed-bitmask mask + per-row max (C layout: VGPR r -> row mb+r, lane n -> col)
    float rmax[8];
#pragma unroll
    for (int r = 0; r < 8; ++r) {
      const unsigned w = pm[(size_t)(i0 + mb + r) * (Nsz / 32) + (j0 >> 5)];
      const float a0 = ((w >> n) & 1u)        ? s0[r] * 0.125f : NEGV;
      const float a1 = ((w >> (16 + n)) & 1u) ? s1[r] * 0.125f : NEGV;
      s0[r] = a0; s1[r] = a1;
      rmax[r] = fmaxf(a0, a1);
    }
#pragma unroll
    for (int off = 1; off < 16; off <<= 1) {
#pragma unroll
      for (int r = 0; r < 8; ++r)
        rmax[r] = fmaxf(rmax[r], __shfl_xor(rmax[r], off, 32));
    }

    // --- online softmax update; write P (bf16) into LDS row-major
    float alpha[8], rsum[8];
#pragma unroll
    for (int r = 0; r < 8; ++r) {
      const float mn = fmaxf(mrow[r], rmax[r]);
      alpha[r] = __expf(mrow[r] - mn);
      mrow[r]  = mn;
      const float p0 = __expf(s0[r] - mn);
      const float p1 = __expf(s1[r] - mn);
      myp[(mb + r) * PSTR + n]      = (__bf16)p0;
      myp[(mb + r) * PSTR + 16 + n] = (__bf16)p1;
      rsum[r] = p0 + p1;
    }
#pragma unroll
    for (int off = 1; off < 16; off <<= 1) {
#pragma unroll
      for (int r = 0; r < 8; ++r)
        rsum[r] += __shfl_xor(rsum[r], off, 32);
    }
#pragma unroll
    for (int r = 0; r < 8; ++r) {
      lrow[r] = lrow[r] * alpha[r] + rsum[r];
      o0[r] *= alpha[r]; o1[r] *= alpha[r]; o2[r] *= alpha[r]; o3[r] *= alpha[r];
    }

    // make LDS P stores visible before re-layout reads (same wave; LDS is in-order)
    asm volatile("s_wait_dscnt 0" ::: "memory");

    // --- read P back as 16x32 A-operand
    v16bf pa;
    {
      const int mm = lane & 15;
      const int ko = (lane & 16) ? 8 : 0;
      const __bf16* rp = myp + mm * PSTR + ko;
      v8bf lo = *(const v8bf*)rp;         // rows are 80B apart -> 16B aligned
      v8bf hi = *(const v8bf*)(rp + 16);
#pragma unroll
      for (int e = 0; e < 8; ++e) { pa[e] = lo[e]; pa[8 + e] = hi[e]; }
    }

    // --- O += P V  (V^T rows contiguous; 4 independent accumulators cover dk=64)
    o0 = wmma_bf16(pa, load_b_bf16(vtb + (size_t)0  * Nsz + j0, Nsz, lane), o0);
    o1 = wmma_bf16(pa, load_b_bf16(vtb + (size_t)16 * Nsz + j0, Nsz, lane), o1);
    o2 = wmma_bf16(pa, load_b_bf16(vtb + (size_t)32 * Nsz + j0, Nsz, lane), o2);
    o3 = wmma_bf16(pa, load_b_bf16(vtb + (size_t)48 * Nsz + j0, Nsz, lane), o3);
  }

  // --- finalize: O /= l, store bf16 in [B,N,D] layout (A-source for out_proj)
  __bf16* ob = (__bf16*)o_p;
#pragma unroll
  for (int r = 0; r < 8; ++r) {
    const int row = i0 + mb + r;
    const float inv = 1.0f / lrow[r];
    __bf16* d = ob + ((size_t)b * Nsz + row) * Dsz + h * DKd;
    d[0 * 16 + n] = (__bf16)(o0[r] * inv);
    d[1 * 16 + n] = (__bf16)(o1[r] * inv);
    d[2 * 16 + n] = (__bf16)(o2[r] * inv);
    d[3 * 16 + n] = (__bf16)(o3[r] * inv);
  }
}

// ============================================================================
// Kernel 3: output projection out = O @ Wo^T (fp32 result), dual accumulators.
// ============================================================================
__global__ void out_proj(const unsigned short* __restrict__ o_p,
                         const unsigned short* __restrict__ wo_p,   // Wo bf16
                         float* __restrict__ out) {
  const int lane = threadIdx.x & 31;
  const int wave = (blockIdx.x * blockDim.x + threadIdx.x) >> 5;
  const int jt = wave & 31;
  const int it = (wave >> 5) & 127;
  const int b  = wave >> 12;
  const int i0 = it * 16, j0 = jt * 16;
  const __bf16* ob = (const __bf16*)o_p + (size_t)b * Nsz * Dsz;
  const __bf16* Wo = (const __bf16*)wo_p;

  v8f c0 = {0.f,0.f,0.f,0.f,0.f,0.f,0.f,0.f};
  v8f c1 = {0.f,0.f,0.f,0.f,0.f,0.f,0.f,0.f};
#pragma unroll 2
  for (int k0 = 0; k0 < Dsz; k0 += 64) {
    c0 = wmma_bf16(load_a_bf16(ob + (size_t)i0 * Dsz + k0,      Dsz, lane),
                   load_b_bf16(Wo + (size_t)j0 * Dsz + k0,      Dsz, lane), c0);
    c1 = wmma_bf16(load_a_bf16(ob + (size_t)i0 * Dsz + k0 + 32, Dsz, lane),
                   load_b_bf16(Wo + (size_t)j0 * Dsz + k0 + 32, Dsz, lane), c1);
  }
#pragma unroll
  for (int r = 0; r < 8; ++r) c0[r] += c1[r];

  const int n  = lane & 15;
  const int mb = (lane & 16) ? 8 : 0;
#pragma unroll
  for (int r = 0; r < 8; ++r)
    out[((size_t)b * Nsz + (i0 + mb + r)) * Dsz + j0 + n] = c0[r];
}

// ============================================================================
extern "C" void kernel_launch(void* const* d_in, const int* in_sizes, int n_in,
                              void* d_out, int out_size, void* d_ws, size_t ws_size,
                              hipStream_t stream) {
  const float* x  = (const float*)d_in[0];
  const float* Wq = (const float*)d_in[1];
  const float* Wk = (const float*)d_in[2];
  const float* Wv = (const float*)d_in[3];
  const float* Wo = (const float*)d_in[4];
  const int* seq_mask     = (const int*)d_in[5];
  const int* sparse_masks = (const int*)d_in[6];
  float* out = (float*)d_out;

  // Workspace layout (24 MiB total):
  //   [0,4)MiB   Q bf16 [B,H,N,64]      [4,8)MiB  K bf16 [B,H,N,64]
  //   [8,12)MiB  V^T bf16 [B,H,64,N]    [12,16)MiB O bf16 [B,N,D]
  //   [16,18)MiB packed masks u32 [B,2,N,N/32]
  //   [18,22)MiB x bf16 [B,N,D]
  //   [22,24)MiB Wq|Wk|Wv|Wo bf16 (4 x 512 KiB)
  char* ws = (char*)d_ws;
  unsigned short* q_ws  = (unsigned short*)(ws + (0ull  << 20));
  unsigned short* k_ws  = (unsigned short*)(ws + (4ull  << 20));
  unsigned short* vt_ws = (unsigned short*)(ws + (8ull  << 20));
  unsigned short* o_ws  = (unsigned short*)(ws + (12ull << 20));
  unsigned*       packd = (unsigned*)      (ws + (16ull << 20));
  unsigned short* x_bf  = (unsigned short*)(ws + (18ull << 20));
  unsigned short* w_bf  = (unsigned short*)(ws + (22ull << 20));

  // 0) pre-convert x and weights to bf16 (removes cvt + halves bytes in GEMM loops)
  cvt_bf16<<<1024, 256, 0, stream>>>(x, x_bf);                       // 2M elems
  cvt_bf16<<<128, 256, 0, stream>>>(Wq, w_bf + 0ull * Dsz * Dsz);    // 256K each
  cvt_bf16<<<128, 256, 0, stream>>>(Wk, w_bf + 1ull * Dsz * Dsz);
  cvt_bf16<<<128, 256, 0, stream>>>(Wv, w_bf + 2ull * Dsz * Dsz);
  cvt_bf16<<<128, 256, 0, stream>>>(Wo, w_bf + 3ull * Dsz * Dsz);

  // 1) pack masks: B*2*N*(N/32) = 524288 words
  pack_masks<<<2048, 256, 0, stream>>>(seq_mask, sparse_masks, packd);
  // 2) QKV projections: 3*B*(N/16)*(D/16) = 24576 waves, 8 waves/block
  qkv_proj<<<3072, 256, 0, stream>>>(x_bf, w_bf, q_ws, k_ws, vt_ws);
  // 3) flash attention: B*H*(N/16) = 2048 waves, 8 waves/block
  flash_attn<<<256, 256, 0, stream>>>(q_ws, k_ws, vt_ws, packd, o_ws);
  // 4) output projection: B*(N/16)*(D/16) = 8192 waves
  out_proj<<<1024, 256, 0, stream>>>(o_ws, w_bf + 3ull * Dsz * Dsz, out);
}
